// AttRNNseq2seq_33097017983192
// MI455X (gfx1250) — matrix operand order repaired and checked by reading, not compiled, not run on hardware
//
#include <hip/hip_runtime.h>
#include <hip/hip_bf16.h>

typedef _Float16 f16;
typedef __attribute__((ext_vector_type(16))) _Float16 v16h;
typedef __attribute__((ext_vector_type(8)))  _Float16 v8h;
typedef __attribute__((ext_vector_type(8)))  float    v8f;

#define BB 32
#define SS 64
#define TT 32
#define EE 300
#define HEE 256
#define HDD 512
#define VDD 50000
#define VDP 50048  // VD padded so tilesN divisible by 4
#define KE 320     // E padded to mult of 32
#define KQ 832     // (E+HD)=812 padded to mult of 32

// d_out layout (floats): logp [B,T,VD], idx [B,T], mask [B,T], att [B,T,S]
static const size_t OUT_LOGP = 0;
static const size_t OUT_IDX  = (size_t)BB*TT*VDD;
static const size_t OUT_MASK = OUT_IDX + (size_t)BB*TT;
static const size_t OUT_ATT  = OUT_MASK + (size_t)BB*TT;

__device__ __forceinline__ float sigm(float x) { return 1.f / (1.f + __expf(-x)); }

// ---------------- WMMA: 1 M-tile x 4 N-tiles, A fragment loaded once per k-step ----
// A: row-major [M][lda] f16 (lda mult of 32).
// B: packed fragment layout: tile (tn, kt) -> 32 lanes x 16 halfs contiguous:
//    offset = ((tn * (K/32) + kt) * 32 + lane) * 16 + j
__device__ __forceinline__ void wmma_tile_acc4(const f16* __restrict__ A, int lda,
                                               const f16* __restrict__ Bp,
                                               int tm, int tn0, int K, int lane,
                                               v8f* __restrict__ c) {
  int m   = tm * 16 + (lane & 15);
  int kbA = (lane >> 4) * 8;   // A: lanes 16-31 hold K base +8 (and +24)
  size_t tileStride = (size_t)(K >> 5) * 512;  // halfs per n-tile of packed B
  const f16* Bt = Bp + (size_t)tn0 * tileStride + (size_t)lane * 16;
  for (int k0 = 0; k0 < K; k0 += 32) {
    const f16* Ap = A + (size_t)m * lda + k0 + kbA;
    v8h alo = *(const v8h*)Ap;          // K = k0+kbA+0..7
    v8h ahi = *(const v8h*)(Ap + 16);   // K = k0+kbA+16..23
    v16h a = __builtin_shufflevector(alo, ahi, 0, 1, 2, 3, 4, 5, 6, 7,
                                     8, 9, 10, 11, 12, 13, 14, 15);
    const f16* Bk = Bt + (size_t)(k0 >> 5) * 512;
#pragma unroll
    for (int j = 0; j < 4; ++j) {
      v16h b = *(const v16h*)(Bk + (size_t)j * tileStride);
      c[j] = __builtin_amdgcn_wmma_f32_16x16x32_f16(false, a, false, b, (short)0,
                                                    c[j], false, false);
    }
  }
}

// Generic GEMM: C[M,N](f32) = A[M,K](f16,row) * B(packed) + bias[N]
// one wave per (M-tile x 4 N-tiles); requires N mult of 64
__global__ void k_gemm4(const f16* __restrict__ A, int lda,
                        const f16* __restrict__ Bp,
                        const float* __restrict__ bias,
                        float* __restrict__ C, int ldc, int M, int N, int K) {
  int wid  = (blockIdx.x * blockDim.x + threadIdx.x) >> 5;
  int lane = threadIdx.x & 31;
  int grpN = N >> 6;  // groups of 4 n-tiles
  int tm = wid / grpN, tn0 = (wid % grpN) * 4;
  if (tm >= (M >> 4)) return;
  v8f c[4] = {};
  wmma_tile_acc4(A, lda, Bp, tm, tn0, K, lane, c);
  int row = tm * 16 + (lane >> 4) * 8;
#pragma unroll
  for (int j = 0; j < 4; ++j) {
    int col = (tn0 + j) * 16 + (lane & 15);
    float bv = bias[col];
#pragma unroll
    for (int v = 0; v < 8; ++v) C[(size_t)(row + v) * ldc + col] = c[j][v] + bv;
  }
}

// ---------------- setup kernels ----------------
// W[Nin][Kin] f32 -> packed WMMA-B fragments [Npad x Kpad] (zero filled)
__global__ void k_convPackB(const float* __restrict__ W, f16* __restrict__ out,
                            int Npad, int Nin, int Kin, int Kpad) {
  size_t idx = (size_t)blockIdx.x * blockDim.x + threadIdx.x;
  size_t total = (size_t)Kpad * Npad;
  if (idx >= total) return;
  int n = (int)(idx % Npad);
  int k = (int)(idx / Npad);
  float v = (k < Kin && n < Nin) ? W[(size_t)n * Kin + k] : 0.f;
  int tn = n >> 4, nl = n & 15;
  int kt = k >> 5, kr = k & 31;
  int hi = (kr >> 3) & 1, j = (kr & 7) + ((kr >> 4) << 3);
  size_t off = ((((size_t)tn * (Kpad >> 5)) + kt) * 32 + (hi * 16 + nl)) * 16 + j;
  out[off] = (f16)v;
}

__global__ void k_padbias(const float* __restrict__ b, float* __restrict__ bp,
                          int Nin, int Npad) {
  int i = blockIdx.x * blockDim.x + threadIdx.x;
  if (i >= Npad) return;
  bp[i] = (i < Nin) ? b[i] : 0.f;
}

// X16[s*B+b][KE] = emb_enc[data[b,s]][e] (pad zero)
__global__ void k_embed_enc(const int* __restrict__ data,
                            const float* __restrict__ emb, f16* __restrict__ X16) {
  int idx = blockIdx.x * blockDim.x + threadIdx.x;
  if (idx >= SS * BB * KE) return;
  int e = idx % KE, sb = idx / KE;
  int b = sb % BB, s = sb / BB;
  int tok = data[b * SS + s];
  X16[idx] = (e < EE) ? (f16)emb[(size_t)tok * EE + e] : (f16)0.f;
}

__global__ void k_init_enc(float* hf, float* hb, f16* hf16, f16* hb16) {
  int i = blockIdx.x * blockDim.x + threadIdx.x;
  if (i >= BB * HEE) return;
  hf[i] = 0.f; hb[i] = 0.f; hf16[i] = (f16)0.f; hb16[i] = (f16)0.f;
}

// ---------------- encoder step: fused [32,256]x[256,768] WMMA + GRU gates ----------
__global__ void k_enc_step(int s, const float* __restrict__ gif,
                           const float* __restrict__ gib,
                           const f16* __restrict__ WhhfP, const f16* __restrict__ WhhbP,
                           const float* __restrict__ bhhf, const float* __restrict__ bhhb,
                           float* hf, float* hb, f16* hf16, f16* hb16,
                           float* ghf, float* ghb, float* __restrict__ enc_out) {
  int dir = blockIdx.x;
  const float* gi  = dir ? gib : gif;
  const f16*  W    = dir ? WhhbP : WhhfP;
  const float* bhh = dir ? bhhb : bhhf;
  float* h  = dir ? hb : hf;
  f16*  hq  = dir ? hb16 : hf16;
  float* gh = dir ? ghb : ghf;
  int srow  = dir ? (SS - 1 - s) : s;   // time index into gi rows / enc_out
  int tid = threadIdx.x, wid = tid >> 5, lane = tid & 31;
  // gh[32,768] = hq[32,256] @ Whh^T[256,768]; 2 x 12 groups of 4 tiles, 32 waves
  for (int t = wid; t < 24; t += 32) {
    int tm = t / 12, tn0 = (t % 12) * 4;
    v8f c[4] = {};
    wmma_tile_acc4(hq, HEE, W, tm, tn0, HEE, lane, c);
    int row = tm * 16 + (lane >> 4) * 8;
#pragma unroll
    for (int j = 0; j < 4; ++j) {
      int col = (tn0 + j) * 16 + (lane & 15);
#pragma unroll
      for (int v = 0; v < 8; ++v) gh[(row + v) * (3 * HEE) + col] = c[j][v];
    }
  }
  __syncthreads();
  for (int i = tid; i < BB * HEE; i += blockDim.x) {
    int b = i >> 8, j = i & 255;
    const float* gir = gi + (size_t)(srow * BB + b) * (3 * HEE);
    float gr = gir[j], gz = gir[HEE + j], gn = gir[2 * HEE + j];
    float hr = gh[b * 3 * HEE + j] + bhh[j];
    float hz = gh[b * 3 * HEE + HEE + j] + bhh[HEE + j];
    float hn = gh[b * 3 * HEE + 2 * HEE + j] + bhh[2 * HEE + j];
    float r = sigm(gr + hr), z = sigm(gz + hz);
    float nn = tanhf(gn + r * hn);
    float hv = (1.f - z) * nn + z * h[i];
    h[i] = hv; hq[i] = (f16)hv;
    enc_out[(size_t)(b * SS + srow) * (2 * HEE) + dir * HEE + j] = hv;
  }
}

// ---------------- decoder ----------------
__global__ void k_dec_init(const float* hf, const float* hb,
                           float* hdec, f16* hdec16, int* inp) {
  int i = blockIdx.x * blockDim.x + threadIdx.x;
  if (i >= BB * HDD) return;
  int b = i >> 9, j = i & 511;
  float v = (j < HEE) ? hf[b * HEE + j] : hb[b * HEE + (j - HEE)];
  hdec[i] = v; hdec16[i] = (f16)v;
  if (i < BB) inp[i] = 2; // SOS
}

__global__ void k_qin(const int* __restrict__ inp, const float* __restrict__ emb_dec,
                      const float* __restrict__ hdec, f16* __restrict__ qin16) {
  int i = blockIdx.x * blockDim.x + threadIdx.x;
  if (i >= BB * KQ) return;
  int b = i / KQ, j = i % KQ;
  float v = 0.f;
  if (j < EE)            v = emb_dec[(size_t)inp[b] * EE + j];
  else if (j < EE + HDD) v = hdec[b * HDD + (j - EE)];
  qin16[i] = (f16)v;
}

__global__ void k_gin(const int* __restrict__ inp, const float* __restrict__ emb_dec,
                      const float* __restrict__ ctx, f16* __restrict__ gin16) {
  int i = blockIdx.x * blockDim.x + threadIdx.x;
  if (i >= BB * KQ) return;
  int b = i / KQ, j = i % KQ;
  float v = 0.f;
  if (j < EE)                v = emb_dec[(size_t)inp[b] * EE + j];
  else if (j < EE + 2 * HEE) v = ctx[b * (2 * HEE) + (j - EE)];
  gin16[i] = (f16)v;
}

// scores -> softmax -> context, one block per batch row
__global__ void k_attn(const float* __restrict__ q, const float* __restrict__ enc_out,
                       float* __restrict__ ctx, float* __restrict__ out, int tstep) {
  int b = blockIdx.x, t = threadIdx.x;
  __shared__ float sc[SS];
  __shared__ float sinv;
  if (t < SS) {
    float s = 0.f;
    const float* e  = enc_out + (size_t)(b * SS + t) * (2 * HEE);
    const float* qb = q + b * (2 * HEE);
    for (int d = 0; d < 2 * HEE; ++d) s += qb[d] * e[d];
    sc[t] = s;
  }
  __syncthreads();
  if (t == 0) {
    float m = sc[0];
    for (int i = 1; i < SS; ++i) m = fmaxf(m, sc[i]);
    float sum = 0.f;
    for (int i = 0; i < SS; ++i) { sc[i] = __expf(sc[i] - m); sum += sc[i]; }
    sinv = 1.f / sum;
  }
  __syncthreads();
  if (t < SS) {
    float a = sc[t] * sinv;
    sc[t] = a;
    out[OUT_ATT + (size_t)(b * TT + tstep) * SS + t] = a;
  }
  __syncthreads();
  for (int d = t; d < 2 * HEE; d += blockDim.x) {
    float s = 0.f;
    for (int i = 0; i < SS; ++i) s += sc[i] * enc_out[(size_t)(b * SS + i) * (2 * HEE) + d];
    ctx[b * (2 * HEE) + d] = s;
  }
}

__global__ void k_gru_dec(const float* __restrict__ gi, const float* __restrict__ gh,
                          float* hdec, f16* hdec16) {
  int i = blockIdx.x * blockDim.x + threadIdx.x;
  if (i >= BB * HDD) return;
  int b = i >> 9, j = i & 511;
  const float* gib = gi + b * 3 * HDD;
  const float* ghb = gh + b * 3 * HDD;
  float r = sigm(gib[j] + ghb[j]);
  float z = sigm(gib[HDD + j] + ghb[HDD + j]);
  float nn = tanhf(gib[2 * HDD + j] + r * ghb[2 * HDD + j]);
  float hv = (1.f - z) * nn + z * hdec[i];
  hdec[i] = hv; hdec16[i] = (f16)hv;
}

// log_softmax + argmax over VD=50000 per batch row; writes logp + idx, feeds back token
__global__ void k_vocab(const float* __restrict__ logits, float* __restrict__ out,
                        int* __restrict__ inp, int tstep) {
  int b = blockIdx.x, t = threadIdx.x;
  __shared__ float smax[256]; __shared__ int sidx[256]; __shared__ float ssum[256];
  const float* lb = logits + (size_t)b * VDP;
  float m = -3.4e38f; int mi = 0;
  for (int v = t; v < VDD; v += 256) {
    float x = lb[v];
    if (x > m) { m = x; mi = v; }
  }
  smax[t] = m; sidx[t] = mi;
  __syncthreads();
  for (int off = 128; off > 0; off >>= 1) {
    if (t < off) {
      if (smax[t + off] > smax[t] ||
          (smax[t + off] == smax[t] && sidx[t + off] < sidx[t])) {
        smax[t] = smax[t + off]; sidx[t] = sidx[t + off];
      }
    }
    __syncthreads();
  }
  float gmax = smax[0]; int gidx = sidx[0];
  float s = 0.f;
  for (int v = t; v < VDD; v += 256) s += __expf(lb[v] - gmax);
  ssum[t] = s;
  __syncthreads();
  for (int off = 128; off > 0; off >>= 1) {
    if (t < off) ssum[t] += ssum[t + off];
    __syncthreads();
  }
  float logZ = gmax + __logf(ssum[0]);
  float* ob = out + OUT_LOGP + ((size_t)b * TT + tstep) * VDD;
  for (int v = t; v < VDD; v += 256) ob[v] = lb[v] - logZ;
  if (t == 0) {
    out[OUT_IDX + b * TT + tstep] = (float)gidx;
    inp[b] = gidx;
  }
}

__global__ void k_mask(float* __restrict__ out) {
  int b = threadIdx.x;
  if (b >= BB) return;
  int cum = 0;
  for (int t = 0; t < TT; ++t) {
    int idx = (int)out[OUT_IDX + b * TT + t];
    if (idx == 3) cum++;
    out[OUT_MASK + b * TT + t] = (cum == 0) ? 1.f : 0.f;
  }
}

// ---------------- host ----------------
extern "C" void kernel_launch(void* const* d_in, const int* in_sizes, int n_in,
                              void* d_out, int out_size, void* d_ws, size_t ws_size,
                              hipStream_t stream) {
  const int*   data    = (const int*)d_in[0];
  const float* emb_enc = (const float*)d_in[4];
  const float* emb_dec = (const float*)d_in[5];
  const float* Wihf = (const float*)d_in[6],  *Whhf = (const float*)d_in[7];
  const float* bihf = (const float*)d_in[8],  *bhhf = (const float*)d_in[9];
  const float* Wihb = (const float*)d_in[10], *Whhb = (const float*)d_in[11];
  const float* bihb = (const float*)d_in[12], *bhhb = (const float*)d_in[13];
  const float* Wihd = (const float*)d_in[14], *Whhd = (const float*)d_in[15];
  const float* bihd = (const float*)d_in[16], *bhhd = (const float*)d_in[17];
  const float* Wattn = (const float*)d_in[18], *battn = (const float*)d_in[19];
  const float* Wout  = (const float*)d_in[20], *bout  = (const float*)d_in[21];
  float* out = (float*)d_out;

  char* p = (char*)d_ws;
  auto alloc = [&](size_t bytes) -> char* {
    char* r = p; p += (bytes + 255) & ~(size_t)255; return r;
  };
  f16* WihfP  = (f16*)alloc((size_t)KE * 768 * 2);
  f16* WihbP  = (f16*)alloc((size_t)KE * 768 * 2);
  f16* WhhfP  = (f16*)alloc((size_t)HEE * 768 * 2);
  f16* WhhbP  = (f16*)alloc((size_t)HEE * 768 * 2);
  f16* WihdP  = (f16*)alloc((size_t)KQ * 1536 * 2);
  f16* WhhdP  = (f16*)alloc((size_t)HDD * 1536 * 2);
  f16* WattnP = (f16*)alloc((size_t)KQ * 512 * 2);
  f16* WoutP  = (f16*)alloc((size_t)HDD * VDP * 2);
  f16* X16    = (f16*)alloc((size_t)SS * BB * KE * 2);
  f16* hf16   = (f16*)alloc((size_t)BB * HEE * 2);
  f16* hb16   = (f16*)alloc((size_t)BB * HEE * 2);
  f16* hdec16 = (f16*)alloc((size_t)BB * HDD * 2);
  f16* qin16  = (f16*)alloc((size_t)BB * KQ * 2);
  f16* gin16  = (f16*)alloc((size_t)BB * KQ * 2);
  float* gif    = (float*)alloc((size_t)SS * BB * 768 * 4);
  float* gib    = (float*)alloc((size_t)SS * BB * 768 * 4);
  float* ghf    = (float*)alloc((size_t)BB * 768 * 4);
  float* ghb    = (float*)alloc((size_t)BB * 768 * 4);
  float* enc_o  = (float*)alloc((size_t)BB * SS * 512 * 4);
  float* hf32   = (float*)alloc((size_t)BB * HEE * 4);
  float* hb32   = (float*)alloc((size_t)BB * HEE * 4);
  float* hdec   = (float*)alloc((size_t)BB * HDD * 4);
  float* qv     = (float*)alloc((size_t)BB * 512 * 4);
  float* ctx    = (float*)alloc((size_t)BB * 512 * 4);
  float* gid    = (float*)alloc((size_t)BB * 1536 * 4);
  float* ghd    = (float*)alloc((size_t)BB * 1536 * 4);
  float* logits = (float*)alloc((size_t)BB * VDP * 4);
  float* boutP  = (float*)alloc((size_t)VDP * 4);
  int*   inp    = (int*)alloc((size_t)BB * 4);

  auto cdiv = [](size_t n, size_t t) { return (int)((n + t - 1) / t); };

  // weight conversion: f32 [Nout][Kin] -> packed WMMA-B f16 fragments
  k_convPackB<<<cdiv((size_t)KE * 768, 256), 256, 0, stream>>>(Wihf, WihfP, 768, 768, EE, KE);
  k_convPackB<<<cdiv((size_t)KE * 768, 256), 256, 0, stream>>>(Wihb, WihbP, 768, 768, EE, KE);
  k_convPackB<<<cdiv((size_t)HEE * 768, 256), 256, 0, stream>>>(Whhf, WhhfP, 768, 768, HEE, HEE);
  k_convPackB<<<cdiv((size_t)HEE * 768, 256), 256, 0, stream>>>(Whhb, WhhbP, 768, 768, HEE, HEE);
  k_convPackB<<<cdiv((size_t)KQ * 1536, 256), 256, 0, stream>>>(Wihd, WihdP, 1536, 1536, 812, KQ);
  k_convPackB<<<cdiv((size_t)HDD * 1536, 256), 256, 0, stream>>>(Whhd, WhhdP, 1536, 1536, HDD, HDD);
  k_convPackB<<<cdiv((size_t)KQ * 512, 256), 256, 0, stream>>>(Wattn, WattnP, 512, 512, 812, KQ);
  k_convPackB<<<cdiv((size_t)HDD * VDP, 256), 256, 0, stream>>>(Wout, WoutP, VDP, VDD, HDD, HDD);
  k_padbias<<<cdiv(VDP, 256), 256, 0, stream>>>(bout, boutP, VDD, VDP);

  // encoder embeddings + batched input projections (hoisted out of the recurrence)
  k_embed_enc<<<cdiv((size_t)SS * BB * KE, 256), 256, 0, stream>>>(data, emb_enc, X16);
  k_init_enc<<<cdiv((size_t)BB * HEE, 256), 256, 0, stream>>>(hf32, hb32, hf16, hb16);
  {
    int waves = (SS * BB / 16) * (768 / 64); // 128 M-tiles x 12 groups
    k_gemm4<<<cdiv((size_t)waves * 32, 256), 256, 0, stream>>>(
        X16, KE, WihfP, bihf, gif, 768, SS * BB, 768, KE);
    k_gemm4<<<cdiv((size_t)waves * 32, 256), 256, 0, stream>>>(
        X16, KE, WihbP, bihb, gib, 768, SS * BB, 768, KE);
  }
  // sequential encoder steps (both directions in parallel as 2 blocks)
  for (int s = 0; s < SS; ++s)
    k_enc_step<<<2, 1024, 0, stream>>>(s, gif, gib, WhhfP, WhhbP, bhhf, bhhb,
                                       hf32, hb32, hf16, hb16, ghf, ghb, enc_o);

  k_dec_init<<<cdiv((size_t)BB * HDD, 256), 256, 0, stream>>>(hf32, hb32, hdec, hdec16, inp);

  for (int t = 0; t < TT; ++t) {
    k_qin<<<cdiv((size_t)BB * KQ, 256), 256, 0, stream>>>(inp, emb_dec, hdec, qin16);
    k_gemm4<<<cdiv((size_t)(2 * 8) * 32, 256), 256, 0, stream>>>(
        qin16, KQ, WattnP, battn, qv, 512, BB, 512, KQ);
    k_attn<<<BB, 256, 0, stream>>>(qv, enc_o, ctx, out, t);
    k_gin<<<cdiv((size_t)BB * KQ, 256), 256, 0, stream>>>(inp, emb_dec, ctx, gin16);
    k_gemm4<<<cdiv((size_t)(2 * 24) * 32, 256), 256, 0, stream>>>(
        gin16, KQ, WihdP, bihd, gid, 1536, BB, 1536, KQ);
    k_gemm4<<<cdiv((size_t)(2 * 24) * 32, 256), 256, 0, stream>>>(
        hdec16, HDD, WhhdP, bhhd, ghd, 1536, BB, 1536, HDD);
    k_gru_dec<<<cdiv((size_t)BB * HDD, 256), 256, 0, stream>>>(gid, ghd, hdec, hdec16);
    k_gemm4<<<cdiv((size_t)(2 * (VDP / 64)) * 32, 256), 256, 0, stream>>>(
        hdec16, HDD, WoutP, boutP, logits, VDP, BB, VDP, HDD);
    k_vocab<<<BB, 256, 0, stream>>>(logits, out, inp, t);
  }
  k_mask<<<1, 32, 0, stream>>>(out);
}
